// MHA_12704513262205
// MI455X (gfx1250) — matrix operand order repaired and checked by reading, not compiled, not run on hardware
//
#include <hip/hip_runtime.h>
#include <hip/hip_bf16.h>

typedef __attribute__((ext_vector_type(16))) __bf16 bf16x16;
typedef __attribute__((ext_vector_type(8)))  __bf16 bf16x8;
typedef __attribute__((ext_vector_type(8)))  float  f32x8;

#define NHEAD  12
#define DMODEL 768
#define HDIM   64
#define SEQ    2048
#define NBATCH 2
#define MROWS  (NBATCH*SEQ)   // 4096

// Async global->LDS DMA (CDNA5): vdst = LDS byte offset (VGPR),
// vaddr = 32-bit byte offset (VGPR), saddr = 64-bit base (SGPR pair).
// Tracked by ASYNCcnt; drained with s_wait_asynccnt.
#define ASYNC_LOAD_B128(ldsoff, voff, base)                        \
  asm volatile("global_load_async_to_lds_b128 %0, %1, %2"          \
               :: "v"(ldsoff), "v"(voff), "s"(base) : "memory")
#define WAIT_ASYNCCNT0() \
  asm volatile("s_wait_asynccnt 0x0" ::: "memory")

// ---------------------------------------------------------------------------
// WMMA fragment helpers (CDNA5 wave32 layouts, per CDNA5 ISA)
// ---------------------------------------------------------------------------

// A-matrix 16x32 bf16: lane -> M = lane%16; K runs [kh*16, kh*16+16)
// with kh = lane/16.
__device__ __forceinline__ bf16x16 frag_A(const __bf16* base, int ld) {
  const int lane = threadIdx.x & 31;
  const int m  = lane & 15;
  const int kh = lane >> 4;
  const __bf16* p = base + m * ld + kh * 16;
  bf16x16 f;
#pragma unroll
  for (int i = 0; i < 8; ++i) f[i] = p[i];
#pragma unroll
  for (int i = 0; i < 8; ++i) f[8 + i] = p[8 + i];
  return f;
}

// B-matrix 32x16 bf16 where B = src^T (src row-major [n][k]); contiguous per lane.
__device__ __forceinline__ bf16x16 frag_BT(const __bf16* base, int ld) {
  const int lane = threadIdx.x & 31;
  const int n  = lane & 15;
  const int kg = lane >> 4;
  const __bf16* p = base + n * ld + kg * 16;
  bf16x16 f;
#pragma unroll
  for (int i = 0; i < 16; ++i) f[i] = p[i];
  return f;
}

// B-matrix 32x16 bf16, src row-major [k][n].
__device__ __forceinline__ bf16x16 frag_B(const __bf16* base, int ld) {
  const int lane = threadIdx.x & 31;
  const int n  = lane & 15;
  const int kg = lane >> 4;
  const __bf16* p = base + (kg * 16) * ld + n;
  bf16x16 f;
#pragma unroll
  for (int i = 0; i < 16; ++i) f[i] = p[i * ld];
  return f;
}

__device__ __forceinline__ f32x8 wmma_bf16(bf16x16 a, bf16x16 b, f32x8 c) {
  return __builtin_amdgcn_wmma_f32_16x16x32_bf16(false, a, false, b,
                                                 (short)0, c, false, false);
}

// ---------------------------------------------------------------------------
// Kernel 1: fused Q/K/V projection (grid.z selects q/k/v)
//   out[b,h,s,hd] = (X @ W^T + bias) * (q ? 1/8 : 1), stored bf16.
//   fp32 inputs are read as b128 and converted to bf16 in-flight.
// ---------------------------------------------------------------------------
__global__ __launch_bounds__(256) void qkv_proj_kernel(
    const float* __restrict__ Xa, const float* __restrict__ Xb,
    const float* __restrict__ Wq, const float* __restrict__ bq,
    const float* __restrict__ Wk, const float* __restrict__ bk,
    const float* __restrict__ Wv, const float* __restrict__ bv,
    __bf16* __restrict__ qo, __bf16* __restrict__ ko, __bf16* __restrict__ vo)
{
  __shared__ __bf16 Xs[128][40];   // 128 rows x 32 K, padded
  __shared__ __bf16 Ws[64][40];    // 64 cols (as rows) x 32 K, padded

  const int z = blockIdx.z;
  const float* X    = (z == 0) ? Xa : Xb;
  const float* W    = (z == 0) ? Wq : (z == 1 ? Wk : Wv);
  const float* bias = (z == 0) ? bq : (z == 1 ? bk : bv);
  __bf16* out       = (z == 0) ? qo : (z == 1 ? ko : vo);
  const float sc    = (z == 0) ? 0.125f : 1.0f;   // fold 1/sqrt(HD) into q

  const int m0   = blockIdx.x * 128;
  const int n0   = blockIdx.y * 64;
  const int tid  = threadIdx.x;
  const int wave = tid >> 5;
  const int lane = tid & 31;
  const int nlo  = lane & 15;
  const int half = lane >> 4;

  f32x8 acc[4] = {};

  for (int k0 = 0; k0 < DMODEL; k0 += 32) {
    // X tile: 1024 float4 chunks (b128 loads), cvt to bf16 in LDS.
    for (int e = tid; e < 128 * 8; e += 256) {
      int r = e >> 3, c = (e & 7) * 4;
      const float4 xv = *(const float4*)&X[(size_t)(m0 + r) * DMODEL + k0 + c];
      Xs[r][c + 0] = (__bf16)xv.x; Xs[r][c + 1] = (__bf16)xv.y;
      Xs[r][c + 2] = (__bf16)xv.z; Xs[r][c + 3] = (__bf16)xv.w;
    }
    // W tile (stored [n][k]): 512 float4 chunks.
    for (int e = tid; e < 64 * 8; e += 256) {
      int r = e >> 3, c = (e & 7) * 4;
      const float4 wv = *(const float4*)&W[(size_t)(n0 + r) * DMODEL + k0 + c];
      Ws[r][c + 0] = (__bf16)wv.x; Ws[r][c + 1] = (__bf16)wv.y;
      Ws[r][c + 2] = (__bf16)wv.z; Ws[r][c + 3] = (__bf16)wv.w;
    }
    __syncthreads();
    bf16x16 a = frag_A(&Xs[wave * 16][0], 40);
#pragma unroll
    for (int j = 0; j < 4; ++j) {
      bf16x16 b = frag_BT(&Ws[j * 16][0], 40);
      acc[j] = wmma_bf16(a, b, acc[j]);
    }
    __syncthreads();
  }

#pragma unroll
  for (int j = 0; j < 4; ++j) {
    int nfull = n0 + j * 16 + nlo;
    int h = nfull >> 6, hd = nfull & 63;
    float bias_n = bias[nfull];
#pragma unroll
    for (int v = 0; v < 8; ++v) {
      int mfull = m0 + wave * 16 + v + 8 * half;
      int bb = mfull >> 11, s = mfull & (SEQ - 1);
      float val = (acc[j][v] + bias_n) * sc;
      out[((size_t)(bb * NHEAD + h) * SEQ + s) * HDIM + hd] = (__bf16)val;
    }
  }
}

// ---------------------------------------------------------------------------
// Kernel 2: flash attention per (b, h, 64 q-rows); never materializes scores.
// K/V 64x64 bf16 tiles are double-buffered in LDS and fetched with
// GLOBAL_LOAD_ASYNC_TO_LDS_B128 (ASYNCcnt), prefetching tile t+1 while
// WMMA + online softmax run on tile t.
// ctx written bf16 in [B, S, D] layout (D index = h*64 + hd).
// ---------------------------------------------------------------------------
__global__ __launch_bounds__(128) void attn_kernel(
    const __bf16* __restrict__ Q, const __bf16* __restrict__ K,
    const __bf16* __restrict__ V, __bf16* __restrict__ ctx)
{
  __shared__ __attribute__((aligned(16))) __bf16 Kt[2][64][72];
  __shared__ __attribute__((aligned(16))) __bf16 Vt[2][64][72];
  __shared__ __attribute__((aligned(16))) __bf16 Ps[4][16][72];

  const int b    = blockIdx.z;
  const int h    = blockIdx.y;
  const int q0   = blockIdx.x * 64;
  const int tid  = threadIdx.x;
  const int wave = tid >> 5;
  const int lane = tid & 31;
  const int nlo  = lane & 15;
  const int half = lane >> 4;

  const size_t head_off = ((size_t)(b * NHEAD + h)) * SEQ * HDIM;
  const __bf16* Qh = Q + head_off;
  const unsigned long long kbase = (unsigned long long)(K + head_off);
  const unsigned long long vbase = (unsigned long long)(V + head_off);

  // Issue async DMA of one 64x64 bf16 K tile + V tile into buffer `buf`.
  // 512 chunks of 16B per matrix; 128 threads -> 4 chunks each.
  auto issue_tile = [&](int kt, int buf) {
#pragma unroll
    for (int i = 0; i < 4; ++i) {
      int chunk = tid + i * 128;
      int r = chunk >> 3;           // 0..63
      int c = (chunk & 7) * 8;      // 0..56, 16B granules
      unsigned voff = (unsigned)(((kt + r) * HDIM + c) * sizeof(__bf16));
      unsigned kl = (unsigned)(size_t)&Kt[buf][r][c];  // LDS offset = addr[31:0]
      unsigned vl = (unsigned)(size_t)&Vt[buf][r][c];
      ASYNC_LOAD_B128(kl, voff, kbase);
      ASYNC_LOAD_B128(vl, voff, vbase);
    }
  };

  // Q fragments for this wave's 16 rows stay resident (K-dim 64 -> 2 frags).
  bf16x16 qf[2];
#pragma unroll
  for (int kk = 0; kk < 2; ++kk)
    qf[kk] = frag_A(Qh + (size_t)(q0 + wave * 16) * HDIM + kk * 32, HDIM);

  f32x8 o[4] = {};
  float mrow[8], lrow[8];
#pragma unroll
  for (int v = 0; v < 8; ++v) { mrow[v] = -1e30f; lrow[v] = 0.0f; }

  // Preload first tile.
  issue_tile(0, 0);
  WAIT_ASYNCCNT0();
  __syncthreads();

  int cur = 0;
  for (int kt = 0; kt < SEQ; kt += 64) {
    // Prefetch next K/V tile into the other buffer while we compute.
    if (kt + 64 < SEQ) issue_tile(kt + 64, cur ^ 1);

    // S = Q K^T  (16 q-rows x 64 keys per wave), fp32 accum.
    f32x8 s[4] = {};
#pragma unroll
    for (int j = 0; j < 4; ++j) {
#pragma unroll
      for (int kk = 0; kk < 2; ++kk) {
        bf16x16 bfr = frag_BT(&Kt[cur][j * 16][kk * 32], 72);
        s[j] = wmma_bf16(qf[kk], bfr, s[j]);
      }
    }

    // Online softmax update. Row (v + 8*half) lives across the 16 lanes of
    // this lane's half-wave; xor masks 1..8 stay inside the half.
#pragma unroll
    for (int v = 0; v < 8; ++v) {
      float mx = fmaxf(fmaxf(s[0][v], s[1][v]), fmaxf(s[2][v], s[3][v]));
#pragma unroll
      for (int d = 1; d < 16; d <<= 1) mx = fmaxf(mx, __shfl_xor(mx, d, 32));
      float mnew  = fmaxf(mrow[v], mx);
      float alpha = __expf(mrow[v] - mnew);
      float rs = 0.0f;
#pragma unroll
      for (int j = 0; j < 4; ++j) {
        float p = __expf(s[j][v] - mnew);
        s[j][v] = p;
        rs += p;
      }
#pragma unroll
      for (int d = 1; d < 16; d <<= 1) rs += __shfl_xor(rs, d, 32);
      lrow[v] = lrow[v] * alpha + rs;
      mrow[v] = mnew;
#pragma unroll
      for (int j = 0; j < 4; ++j) o[j][v] = o[j][v] * alpha;
    }

    // Re-layout P (C-layout) -> A-layout through per-wave LDS slab.
#pragma unroll
    for (int j = 0; j < 4; ++j)
#pragma unroll
      for (int v = 0; v < 8; ++v)
        Ps[wave][v + 8 * half][j * 16 + nlo] = (__bf16)s[j][v];

    // O += P V  (K-dim 64 -> 2 steps).
#pragma unroll
    for (int kk = 0; kk < 2; ++kk) {
      bf16x16 pf = frag_A(&Ps[wave][0][kk * 32], 72);
#pragma unroll
      for (int j = 0; j < 4; ++j) {
        bf16x16 vf = frag_B(&Vt[cur][kk * 32][j * 16], 72);
        o[j] = wmma_bf16(pf, vf, o[j]);
      }
    }

    // Drain this wave's prefetch, then rendezvous before buffer swap.
    WAIT_ASYNCCNT0();
    __syncthreads();
    cur ^= 1;
  }

  // Normalize and store ctx (bf16, [B,S,D] with D = h*64 + hd).
#pragma unroll
  for (int j = 0; j < 4; ++j) {
#pragma unroll
    for (int v = 0; v < 8; ++v) {
      int qrow = q0 + wave * 16 + v + 8 * half;
      float val = o[j][v] / lrow[v];
      ctx[((size_t)(b * SEQ + qrow)) * DMODEL + h * HDIM + j * 16 + nlo] = (__bf16)val;
    }
  }
}

// ---------------------------------------------------------------------------
// Kernel 3: output projection  proj = ctx @ Wd^T + bd  (fp32 out)
// ---------------------------------------------------------------------------
__global__ __launch_bounds__(256) void out_proj_kernel(
    const __bf16* __restrict__ ctx, const float* __restrict__ Wd,
    const float* __restrict__ bd, float* __restrict__ proj)
{
  __shared__ __attribute__((aligned(16))) __bf16 As[128][40];
  __shared__ __bf16 Ws[64][40];

  const int m0   = blockIdx.x * 128;
  const int n0   = blockIdx.y * 64;
  const int tid  = threadIdx.x;
  const int wave = tid >> 5;
  const int lane = tid & 31;
  const int nlo  = lane & 15;
  const int half = lane >> 4;

  f32x8 acc[4] = {};

  for (int k0 = 0; k0 < DMODEL; k0 += 32) {
    // ctx tile is already bf16: copy in 16B (8 x bf16) chunks.
    for (int e = tid; e < 128 * 4; e += 256) {
      int r = e >> 2, c = (e & 3) * 8;
      const bf16x8 av = *(const bf16x8*)&ctx[(size_t)(m0 + r) * DMODEL + k0 + c];
#pragma unroll
      for (int u = 0; u < 8; ++u) As[r][c + u] = av[u];
    }
    // W tile (stored [n][k]): float4 loads + cvt.
    for (int e = tid; e < 64 * 8; e += 256) {
      int r = e >> 3, c = (e & 7) * 4;
      const float4 wv = *(const float4*)&Wd[(size_t)(n0 + r) * DMODEL + k0 + c];
      Ws[r][c + 0] = (__bf16)wv.x; Ws[r][c + 1] = (__bf16)wv.y;
      Ws[r][c + 2] = (__bf16)wv.z; Ws[r][c + 3] = (__bf16)wv.w;
    }
    __syncthreads();
    bf16x16 a = frag_A(&As[wave * 16][0], 40);
#pragma unroll
    for (int j = 0; j < 4; ++j) {
      bf16x16 bfr = frag_BT(&Ws[j * 16][0], 40);
      acc[j] = wmma_bf16(a, bfr, acc[j]);
    }
    __syncthreads();
  }

#pragma unroll
  for (int j = 0; j < 4; ++j) {
    float bias = bd[n0 + j * 16 + nlo];
#pragma unroll
    for (int v = 0; v < 8; ++v) {
      int m = m0 + wave * 16 + v + 8 * half;
      proj[(size_t)m * DMODEL + n0 + j * 16 + nlo] = acc[j][v] + bias;
    }
  }
}

// ---------------------------------------------------------------------------
// Kernel 4: residual + LayerNorm, one block (256 thr) per row of 768.
// ---------------------------------------------------------------------------
__global__ __launch_bounds__(256) void ln_kernel(
    const float* __restrict__ Xa, const float* __restrict__ proj,
    const float* __restrict__ gamma, const float* __restrict__ beta,
    float* __restrict__ out)
{
  __shared__ float red[16];
  const int row = blockIdx.x;
  const int tid = threadIdx.x;

  float x[3];
  float sum = 0.0f;
#pragma unroll
  for (int i = 0; i < 3; ++i) {
    int c = tid + i * 256;
    x[i] = Xa[(size_t)row * DMODEL + c] + proj[(size_t)row * DMODEL + c];
    sum += x[i];
  }
  for (int d = 1; d < 32; d <<= 1) sum += __shfl_xor(sum, d, 32);
  if ((tid & 31) == 0) red[tid >> 5] = sum;
  __syncthreads();
  if (tid == 0) { float t = 0.f; for (int i = 0; i < 8; ++i) t += red[i]; red[8] = t; }
  __syncthreads();
  const float mu = red[8] * (1.0f / (float)DMODEL);

  float vs = 0.0f;
#pragma unroll
  for (int i = 0; i < 3; ++i) { float d = x[i] - mu; vs += d * d; }
  for (int d = 1; d < 32; d <<= 1) vs += __shfl_xor(vs, d, 32);
  __syncthreads();
  if ((tid & 31) == 0) red[tid >> 5] = vs;
  __syncthreads();
  if (tid == 0) { float t = 0.f; for (int i = 0; i < 8; ++i) t += red[i]; red[8] = t; }
  __syncthreads();
  const float rstd = rsqrtf(red[8] * (1.0f / (float)DMODEL) + 1e-12f);

#pragma unroll
  for (int i = 0; i < 3; ++i) {
    int c = tid + i * 256;
    out[(size_t)row * DMODEL + c] = (x[i] - mu) * rstd * gamma[c] + beta[c];
  }
}

// ---------------------------------------------------------------------------
extern "C" void kernel_launch(void* const* d_in, const int* in_sizes, int n_in,
                              void* d_out, int out_size, void* d_ws, size_t ws_size,
                              hipStream_t stream)
{
  const float* Xa    = (const float*)d_in[0];
  const float* Xb    = (const float*)d_in[1];
  const float* Wq    = (const float*)d_in[2];
  const float* bq    = (const float*)d_in[3];
  const float* Wk    = (const float*)d_in[4];
  const float* bk    = (const float*)d_in[5];
  const float* Wv    = (const float*)d_in[6];
  const float* bv    = (const float*)d_in[7];
  const float* Wd    = (const float*)d_in[8];
  const float* bd    = (const float*)d_in[9];
  const float* gamma = (const float*)d_in[10];
  const float* beta  = (const float*)d_in[11];
  float* out = (float*)d_out;

  char* ws = (char*)d_ws;
  const size_t headElems = (size_t)NBATCH * NHEAD * SEQ * HDIM;   // 3.1M
  __bf16* qb  = (__bf16*)ws; ws += headElems * sizeof(__bf16);
  __bf16* kb  = (__bf16*)ws; ws += headElems * sizeof(__bf16);
  __bf16* vb  = (__bf16*)ws; ws += headElems * sizeof(__bf16);
  __bf16* ctx = (__bf16*)ws; ws += (size_t)MROWS * DMODEL * sizeof(__bf16);
  float*  proj = (float*)ws;

  qkv_proj_kernel<<<dim3(MROWS / 128, DMODEL / 64, 3), 256, 0, stream>>>(
      Xa, Xb, Wq, bq, Wk, bk, Wv, bv, qb, kb, vb);
  attn_kernel<<<dim3(SEQ / 64, NHEAD, NBATCH), 128, 0, stream>>>(qb, kb, vb, ctx);
  out_proj_kernel<<<dim3(MROWS / 128, DMODEL / 64), 256, 0, stream>>>(ctx, Wd, bd, proj);
  ln_kernel<<<MROWS, 256, 0, stream>>>(Xa, proj, gamma, beta, out);
}
